// P4TransformerDA5_27788438405208
// MI455X (gfx1250) — matrix operand order repaired and checked by reading, not compiled, not run on hardware
//
#include <hip/hip_runtime.h>
#include <hip/hip_bf16.h>
#include <cstdint>

// ---------------- model constants ----------------
#define B_    4
#define L_    8
#define N_    1024
#define P_    64
#define NT_   1088      // N_ + P_
#define M_    136       // NT_/8
#define KNB   32        // ball-query K
#define DIM_  128
#define HEADS_ 8
#define DH_   16
#define MLP_  256
#define HP_   4
#define DHP_  32
#define J_    13
#define MEMN_ 1024
#define FRAMES_ 32      // B_*L_
#define RT_   4352      // B_*L_*M_ == B_*NT_
#define RADIUS2 0.25f

typedef __attribute__((ext_vector_type(16))) __bf16 v16bf;
typedef __attribute__((ext_vector_type(8)))  __bf16 v8bf;
typedef __attribute__((ext_vector_type(8)))  float  v8f;
typedef __attribute__((ext_vector_type(4)))  unsigned int u32x4;
typedef __attribute__((ext_vector_type(8)))  int    i32x8;
typedef __attribute__((ext_vector_type(4)))  int    i32x4;

__device__ __forceinline__ float gelu_f(float x) {
    return 0.5f * x * (1.0f + erff(x * 0.70710678118654752f));
}
__device__ __forceinline__ int iclamp(int v, int lo, int hi) {
    return v < lo ? lo : (v > hi ? hi : v);
}

#define GTILE 32
#define LDSPAD 8

// ---------------- shared WMMA tile epilogue helpers ----------------
__device__ __forceinline__ float apply_act(float v, int act) {
    if (act == 1) return gelu_f(v);
    if (act == 2) return fmaxf(v, 0.0f);
    return v;
}

// =====================================================================
// FAST batched WMMA GEMM: interior-only. Requires Mr,Nc,Kd % 32 == 0 and
// lda,ldb % 8 == 0 (16B-aligned rows). Grid covers the matrix exactly.
// One v8bf global load + one b128 LDS store per thread per tile; no guards.
// =====================================================================
template<int TRANSB>
__global__ __launch_bounds__(128)
void gemm_wmma_fast(const __bf16* __restrict__ A, int lda, long sA,
                    const __bf16* __restrict__ Bp, int ldb, long sB,
                    void* __restrict__ Cout, int ldc, long sC,
                    int Kd, const float* __restrict__ bias,
                    float alpha, int act, int outBf16)
{
    __shared__ __bf16 As[GTILE][GTILE + LDSPAD];
    __shared__ __bf16 Bs[GTILE][GTILE + LDSPAD];   // Bs[n][k]

    const int z    = blockIdx.z;
    const __bf16* Ab = A  + (long)z * sA;
    const __bf16* Bb = Bp + (long)z * sB;

    const int tid  = threadIdx.x;
    const int wave = tid >> 5;
    const int lane = tid & 31;
    const int l15  = lane & 15;
    const int half = lane >> 4;
    const int rowBase = (wave >> 1) << 4;
    const int colBase = (wave & 1) << 4;
    const int blockM = blockIdx.y * GTILE;
    const int blockN = blockIdx.x * GTILE;

    const int sr = tid >> 2;           // 0..31
    const int sc = (tid & 3) << 3;     // 0,8,16,24

    v8f acc = {};

    for (int k0 = 0; k0 < Kd; k0 += GTILE) {
        // stage A: thread t -> row blockM+sr, k chunk k0+sc (16B)
        *(v8bf*)&As[sr][sc] = *(const v8bf*)&Ab[(long)(blockM + sr) * lda + (k0 + sc)];
        if (k0 + GTILE < Kd)
            __builtin_prefetch((const void*)&Ab[(long)(blockM + sr) * lda + (k0 + GTILE + sc)], 0, 1);
        // stage B transposed into Bs[n][k]
        if (TRANSB) {
            *(v8bf*)&Bs[sr][sc] = *(const v8bf*)&Bb[(long)(blockN + sr) * ldb + (k0 + sc)];
        } else {
            v8bf v = *(const v8bf*)&Bb[(long)(k0 + sr) * ldb + (blockN + sc)];
            #pragma unroll
            for (int j = 0; j < 8; ++j) Bs[sc + j][sr] = v[j];
        }
        __syncthreads();

        const int kbA = half * 8;
        v8bf alo = *(const v8bf*)&As[rowBase + l15][kbA];
        v8bf ahi = *(const v8bf*)&As[rowBase + l15][kbA + 16];
        v16bf af = __builtin_shufflevector(alo, ahi, 0,1,2,3,4,5,6,7,8,9,10,11,12,13,14,15);
        const int kbB = half * 8;
        v8bf blo = *(const v8bf*)&Bs[colBase + l15][kbB];
        v8bf bhi = *(const v8bf*)&Bs[colBase + l15][kbB + 16];
        v16bf bfv = __builtin_shufflevector(blo, bhi, 0,1,2,3,4,5,6,7,8,9,10,11,12,13,14,15);

        acc = __builtin_amdgcn_wmma_f32_16x16x32_bf16(false, af, false, bfv,
                                                      (short)0, acc, false, false);
        __syncthreads();
    }

    const int n  = blockN + colBase + l15;
    const int m0 = blockM + rowBase + half * 8;
    const float bval = bias ? bias[n] : 0.0f;
    if (outBf16) {
        __bf16* Cp = (__bf16*)Cout + (long)z * sC;
        #pragma unroll
        for (int e = 0; e < 8; ++e)
            Cp[(long)(m0 + e) * ldc + n] = (__bf16)apply_act(acc[e] * alpha + bval, act);
    } else {
        float* Cp = (float*)Cout + (long)z * sC;
        #pragma unroll
        for (int e = 0; e < 8; ++e)
            Cp[(long)(m0 + e) * ldc + n] = apply_act(acc[e] * alpha + bval, act);
    }
}

// =====================================================================
// GENERIC batched WMMA GEMM: arbitrary shapes; clamped-index staging
// (branch-free cndmask zero-fill), guarded epilogue.
// =====================================================================
__global__ __launch_bounds__(128)
void gemm_wmma_gen(const __bf16* __restrict__ A, int lda, long sA,
                   const __bf16* __restrict__ Bp, int ldb, long sB,
                   void* __restrict__ Cout, int ldc, long sC,
                   int Mr, int Nc, int Kd,
                   const float* __restrict__ bias,
                   float alpha, int act, int transB, int outBf16)
{
    __shared__ __bf16 As[GTILE][GTILE + LDSPAD];
    __shared__ __bf16 Bs[GTILE][GTILE + LDSPAD];   // Bs[n][k]

    const int z    = blockIdx.z;
    const __bf16* Ab = A  + (long)z * sA;
    const __bf16* Bb = Bp + (long)z * sB;

    const int tid  = threadIdx.x;
    const int wave = tid >> 5;
    const int lane = tid & 31;
    const int l15  = lane & 15;
    const int half = lane >> 4;
    const int rowBase = (wave >> 1) << 4;
    const int colBase = (wave & 1) << 4;
    const int blockM = blockIdx.y * GTILE;
    const int blockN = blockIdx.x * GTILE;

    v8f acc = {};

    for (int k0 = 0; k0 < Kd; k0 += GTILE) {
        #pragma unroll
        for (int j = 0; j < 8; ++j) {
            int e  = tid * 8 + j;
            int r  = e >> 5, kk = e & 31;
            int gr = blockM + r, gk = k0 + kk;
            bool ok = (gr < Mr) && (gk < Kd);
            long ai = ok ? ((long)gr * lda + gk) : 0;   // clamped, always valid
            __bf16 v = Ab[ai];
            As[r][kk] = ok ? v : (__bf16)0.0f;
        }
        #pragma unroll
        for (int j = 0; j < 8; ++j) {
            int e  = tid * 8 + j;
            int n  = e >> 5, kk = e & 31;
            int gn = blockN + n, gk = k0 + kk;
            bool ok = (gn < Nc) && (gk < Kd);
            long bi = ok ? (transB ? ((long)gn * ldb + gk) : ((long)gk * ldb + gn)) : 0;
            __bf16 v = Bb[bi];
            Bs[n][kk] = ok ? v : (__bf16)0.0f;
        }
        __syncthreads();

        const int kbA = half * 8;
        v8bf alo = *(const v8bf*)&As[rowBase + l15][kbA];
        v8bf ahi = *(const v8bf*)&As[rowBase + l15][kbA + 16];
        v16bf af = __builtin_shufflevector(alo, ahi, 0,1,2,3,4,5,6,7,8,9,10,11,12,13,14,15);
        const int kbB = half * 8;
        v8bf blo = *(const v8bf*)&Bs[colBase + l15][kbB];
        v8bf bhi = *(const v8bf*)&Bs[colBase + l15][kbB + 16];
        v16bf bfv = __builtin_shufflevector(blo, bhi, 0,1,2,3,4,5,6,7,8,9,10,11,12,13,14,15);

        acc = __builtin_amdgcn_wmma_f32_16x16x32_bf16(false, af, false, bfv,
                                                      (short)0, acc, false, false);
        __syncthreads();
    }

    const int n  = blockN + colBase + l15;
    const int m0 = blockM + rowBase + half * 8;
    const bool nok = n < Nc;
    const float bval = (bias && nok) ? bias[n] : 0.0f;
    #pragma unroll
    for (int e = 0; e < 8; ++e) {
        int m = m0 + e;
        if (nok && m < Mr) {
            float v = apply_act(acc[e] * alpha + bval, act);
            if (outBf16) ((__bf16*)Cout)[(long)z * sC + (long)m * ldc + n] = (__bf16)v;
            else         ((float*) Cout)[(long)z * sC + (long)m * ldc + n] = v;
        }
    }
}

// ---------------- f32 -> bf16 convert ----------------
__global__ void convert_kernel(const float* __restrict__ x, __bf16* __restrict__ y, int n) {
    int i = blockIdx.x * 256 + threadIdx.x;
    if (i < n) y[i] = (__bf16)x[i];
}

// ---------------- LayerNorm (D=128), output bf16, optional fused GELU ----------------
__global__ __launch_bounds__(128)
void ln_kernel(const float* __restrict__ X, const float* __restrict__ g,
               const float* __restrict__ bta, __bf16* __restrict__ Y, int applyGelu)
{
    const size_t row = blockIdx.x;
    const float* x = X + row * DIM_;
    __shared__ float red[DIM_];
    int tid = threadIdx.x;
    float v = x[tid];
    red[tid] = v; __syncthreads();
    for (int s = 64; s > 0; s >>= 1) { if (tid < s) red[tid] += red[tid + s]; __syncthreads(); }
    float mean = red[0] * (1.0f / DIM_); __syncthreads();
    float dv = v - mean;
    red[tid] = dv * dv; __syncthreads();
    for (int s = 64; s > 0; s >>= 1) { if (tid < s) red[tid] += red[tid + s]; __syncthreads(); }
    float var = red[0] * (1.0f / DIM_);
    float o = dv * rsqrtf(var + 1e-5f) * g[tid] + bta[tid];
    if (applyGelu) o = gelu_f(o);
    Y[row * DIM_ + tid] = (__bf16)o;
}

// ---------------- row softmax -> bf16 ----------------
__global__ __launch_bounds__(256)
void softmax_kernel(const float* __restrict__ S, __bf16* __restrict__ Pb, int cols, int ldin)
{
    const size_t row = blockIdx.x;
    const float* x = S + row * (size_t)ldin;
    __bf16* y = Pb + row * (size_t)cols;
    __shared__ float red[256];
    int tid = threadIdx.x;
    float mx = -3.4e38f;
    for (int j = tid; j < cols; j += 256) mx = fmaxf(mx, x[j]);
    red[tid] = mx; __syncthreads();
    for (int s = 128; s > 0; s >>= 1) { if (tid < s) red[tid] = fmaxf(red[tid], red[tid + s]); __syncthreads(); }
    mx = red[0]; __syncthreads();
    float sum = 0.f;
    for (int j = tid; j < cols; j += 256) sum += __expf(x[j] - mx);
    red[tid] = sum; __syncthreads();
    for (int s = 128; s > 0; s >>= 1) { if (tid < s) red[tid] += red[tid + s]; __syncthreads(); }
    float inv = 1.0f / red[0];
    for (int j = tid; j < cols; j += 256) y[j] = (__bf16)(__expf(x[j] - mx) * inv);
}

// ---------------- elementwise ----------------
__global__ void add_inplace_kernel(float* __restrict__ x, const float* __restrict__ y, int n) {
    int i = blockIdx.x * 256 + threadIdx.x;
    if (i < n) x[i] += y[i];
}
__global__ void add_relu_kernel(float* __restrict__ x, const float* __restrict__ y, int n) {
    int i = blockIdx.x * 256 + threadIdx.x;
    if (i < n) x[i] = fmaxf(x[i] + y[i], 0.0f);
}
__global__ void scalebias_relu_kernel(const float* __restrict__ x, const float* __restrict__ g,
                                      const float* __restrict__ b, float* __restrict__ y, int n) {
    int i = blockIdx.x * 256 + threadIdx.x;
    if (i < n) {
        int c = i & (DIM_ - 1);
        y[i] = fmaxf(x[i] * (g[c] * 0.999995f) + b[c], 0.0f);
    }
}

// ---------------- assemble full = concat(flat, prop) ----------------
__global__ void build_full_kernel(const float* __restrict__ flat, const float* __restrict__ prop,
                                  float* __restrict__ full) {
    int t = blockIdx.x * 256 + threadIdx.x;
    const int total = FRAMES_ * NT_ * 6;
    if (t >= total) return;
    int c = t % 6; int rest = t / 6; int p = rest % NT_; int f = rest / NT_;
    float v;
    if (p < N_) v = flat[((size_t)f * N_ + p) * 6 + c];
    else        v = prop[((size_t)f * P_ + (p - N_)) * 6 + c];
    full[t] = v;
}

// ---------------- farthest point sampling, one block per frame ----------------
__global__ __launch_bounds__(256)
void fps_kernel(const float* __restrict__ full, int* __restrict__ idx)
{
    const int f = blockIdx.x;
    const float* pts = full + (size_t)f * NT_ * 6;
    __shared__ float dist[NT_];
    __shared__ float rv[256];
    __shared__ int   ri[256];
    __shared__ int   lastIdx;
    int tid = threadIdx.x;
    for (int j = tid; j < NT_; j += 256) dist[j] = 1e10f;
    if (tid == 0) { idx[(size_t)f * M_] = 0; lastIdx = 0; }
    __syncthreads();
    for (int i = 1; i < M_; ++i) {
        int li = lastIdx;
        float px = pts[li * 6], py = pts[li * 6 + 1], pz = pts[li * 6 + 2];
        float bv = -1.0f; int bi = NT_;
        for (int j = tid; j < NT_; j += 256) {
            float dx = pts[j * 6] - px, dy = pts[j * 6 + 1] - py, dz = pts[j * 6 + 2] - pz;
            float dd = fminf(dist[j], dx * dx + dy * dy + dz * dz);
            dist[j] = dd;
            if (dd > bv || (dd == bv && j < bi)) { bv = dd; bi = j; }
        }
        rv[tid] = bv; ri[tid] = bi;
        __syncthreads();
        for (int s = 128; s > 0; s >>= 1) {
            if (tid < s) {
                if (rv[tid + s] > rv[tid] || (rv[tid + s] == rv[tid] && ri[tid + s] < ri[tid])) {
                    rv[tid] = rv[tid + s]; ri[tid] = ri[tid + s];
                }
            }
            __syncthreads();
        }
        if (tid == 0) { lastIdx = ri[0]; idx[(size_t)f * M_ + i] = ri[0]; }
        __syncthreads();
    }
}

// ---------------- gather anchor xyz ----------------
__global__ void gather_anchor_kernel(const float* __restrict__ full, const int* __restrict__ idx,
                                     float* __restrict__ anchor) {
    int t = blockIdx.x * 256 + threadIdx.x;
    if (t >= FRAMES_ * M_) return;
    int f = t / M_;
    int j = idx[t];
    const float* p = full + ((size_t)f * NT_ + j) * 6;
    anchor[(size_t)t * 3 + 0] = p[0];
    anchor[(size_t)t * 3 + 1] = p[1];
    anchor[(size_t)t * 3 + 2] = p[2];
}

// ---------------- ball query + tube embedding max-pool (one block per (frame,m)) ----------------
__global__ __launch_bounds__(128)
void tube_kernel(const float* __restrict__ full, const float* __restrict__ anchor,
                 const float* __restrict__ Wd, const float* __restrict__ Wf,
                 float* __restrict__ pooled, int offset, int isFirst)
{
    const int m = blockIdx.x;
    const int f = blockIdx.y;
    const int b = f / L_, l = f % L_;
    const int lc = iclamp(l + offset, 0, L_ - 1);
    const float* pts = full + ((size_t)(b * L_ + lc)) * NT_ * 6;

    __shared__ float sWd[4 * DIM_];
    __shared__ float sWf[3 * DIM_];
    __shared__ int   nbr[KNB];
    __shared__ float npt[KNB][6];
    __shared__ float anc[3];
    int tid = threadIdx.x;
    for (int i = tid; i < 4 * DIM_; i += 128) sWd[i] = Wd[i];
    for (int i = tid; i < 3 * DIM_; i += 128) sWf[i] = Wf[i];
    if (tid < 3) anc[tid] = anchor[((size_t)f * M_ + m) * 3 + tid];
    __syncthreads();
    if (tid == 0) {
        int cnt = 0;
        float ax = anc[0], ay = anc[1], az = anc[2];
        for (int j = 0; j < NT_ && cnt < KNB; ++j) {
            float dx = pts[j * 6] - ax, dy = pts[j * 6 + 1] - ay, dz = pts[j * 6 + 2] - az;
            if (dx * dx + dy * dy + dz * dz < RADIUS2) nbr[cnt++] = j;
        }
        int pad = (cnt == 0) ? 0 : nbr[0];
        for (int j = cnt; j < KNB; ++j) nbr[j] = pad;
    }
    __syncthreads();
    for (int i = tid; i < KNB * 6; i += 128) {
        int kk = i / 6, c = i % 6;
        npt[kk][c] = pts[(size_t)nbr[kk] * 6 + c];
    }
    __syncthreads();
    const int d = tid;
    const float fo = (float)offset;
    float best = -3.4e38f;
    for (int kk = 0; kk < KNB; ++kk) {
        float dx = npt[kk][0] - anc[0], dy = npt[kk][1] - anc[1], dz = npt[kk][2] - anc[2];
        float s = dx * sWd[d] + dy * sWd[DIM_ + d] + dz * sWd[2 * DIM_ + d] + fo * sWd[3 * DIM_ + d]
                + npt[kk][3] * sWf[d] + npt[kk][4] * sWf[DIM_ + d] + npt[kk][5] * sWf[2 * DIM_ + d];
        best = fmaxf(best, s);
    }
    size_t oi = ((size_t)f * M_ + m) * DIM_ + d;
    pooled[oi] = isFirst ? best : fmaxf(pooled[oi], best);
}

// ---------------- build xyzts (anchor + time) ----------------
__global__ void xyzts_kernel(const float* __restrict__ anchor, float* __restrict__ xyzts) {
    int t = blockIdx.x * 256 + threadIdx.x;
    if (t >= FRAMES_ * M_) return;
    int f = t / M_; int l = f % L_;
    xyzts[(size_t)t * 4 + 0] = anchor[(size_t)t * 3 + 0];
    xyzts[(size_t)t * 4 + 1] = anchor[(size_t)t * 3 + 1];
    xyzts[(size_t)t * 4 + 2] = anchor[(size_t)t * 3 + 2];
    xyzts[(size_t)t * 4 + 3] = (float)(l + 1);
}

// ---------------- 3-NN interpolation at mid frame -> deccat (bf16, 131 cols) ----------------
__global__ void three_interp_kernel(const float* __restrict__ full, const float* __restrict__ anchor,
                                    const float* __restrict__ feats, __bf16* __restrict__ deccat)
{
    int t = blockIdx.x * 256 + threadIdx.x;
    if (t >= B_ * NT_) return;
    int b = t / NT_, n = t % NT_;
    int f = b * L_ + L_ / 2;
    const float* px  = full + ((size_t)f * NT_ + n) * 6;
    const float* anc = anchor + (size_t)f * M_ * 3;
    float d0 = 3e38f, d1 = 3e38f, d2 = 3e38f; int i0 = 0, i1 = 0, i2 = 0;
    for (int mm = 0; mm < M_; ++mm) {
        float dx = px[0] - anc[mm * 3], dy = px[1] - anc[mm * 3 + 1], dz = px[2] - anc[mm * 3 + 2];
        float dd = dx * dx + dy * dy + dz * dz;
        if (dd < d0)      { d2 = d1; i2 = i1; d1 = d0; i1 = i0; d0 = dd; i0 = mm; }
        else if (dd < d1) { d2 = d1; i2 = i1; d1 = dd; i1 = mm; }
        else if (dd < d2) { d2 = dd; i2 = mm; }
    }
    float w0 = 1.0f / (d0 + 1e-8f), w1 = 1.0f / (d1 + 1e-8f), w2 = 1.0f / (d2 + 1e-8f);
    float ws = 1.0f / (w0 + w1 + w2);
    w0 *= ws; w1 *= ws; w2 *= ws;
    const size_t fb = (size_t)b * (L_ * M_) + (size_t)(L_ / 2) * M_;
    const float* f0 = feats + (fb + i0) * DIM_;
    const float* f1 = feats + (fb + i1) * DIM_;
    const float* f2 = feats + (fb + i2) * DIM_;
    __bf16* out = deccat + (size_t)t * 131;
    for (int c = 0; c < DIM_; ++c) out[c] = (__bf16)(w0 * f0[c] + w1 * f1[c] + w2 * f2[c]);
    out[128] = (__bf16)px[3]; out[129] = (__bf16)px[4]; out[130] = (__bf16)px[5];
}

// ---------------- head: softmax over points + weighted xyz sum ----------------
__global__ __launch_bounds__(256)
void head_final_kernel(const float* __restrict__ logits, const float* __restrict__ full,
                       float* __restrict__ out)
{
    int bj = blockIdx.x; int b = bj / J_;
    const float* lg  = logits + (size_t)b * NT_ * J_ + (bj % J_);
    const float* pts = full + (size_t)(b * L_ + L_ / 2) * NT_ * 6;
    __shared__ float red[256];
    int tid = threadIdx.x;
    float mx = -3.4e38f;
    for (int n = tid; n < NT_; n += 256) mx = fmaxf(mx, lg[(size_t)n * J_]);
    red[tid] = mx; __syncthreads();
    for (int s = 128; s > 0; s >>= 1) { if (tid < s) red[tid] = fmaxf(red[tid], red[tid + s]); __syncthreads(); }
    mx = red[0]; __syncthreads();
    float se = 0, sx = 0, sy = 0, sz = 0;
    for (int n = tid; n < NT_; n += 256) {
        float e = __expf(lg[(size_t)n * J_] - mx);
        se += e; sx += e * pts[n * 6]; sy += e * pts[n * 6 + 1]; sz += e * pts[n * 6 + 2];
    }
    float vals[4] = { se, sx, sy, sz };
    for (int q = 0; q < 4; ++q) {
        red[tid] = vals[q]; __syncthreads();
        for (int s = 128; s > 0; s >>= 1) { if (tid < s) red[tid] += red[tid + s]; __syncthreads(); }
        vals[q] = red[0]; __syncthreads();
    }
    if (tid == 0) {
        float inv = 1.0f / vals[0];
        out[(size_t)bj * 3 + 0] = vals[1] * inv;
        out[(size_t)bj * 3 + 1] = vals[2] * inv;
        out[(size_t)bj * 3 + 2] = vals[3] * inv;
    }
}

// ---------------- Tensor Data Mover warm-up: async tile DMA into LDS ----------------
__global__ __launch_bounds__(256)
void tdm_warm_kernel(const float* __restrict__ src, float* __restrict__ dst)
{
    __shared__ float buf[1024];
    unsigned long long ga = (unsigned long long)(size_t)src;
    u32x4 g0;
    g0[0] = 1u;                                             // count = 1 valid descriptor
    g0[1] = 0u;                                             // lds_addr = 0
    g0[2] = (unsigned)(ga & 0xFFFFFFFFull);                 // global_addr[31:0]
    g0[3] = (unsigned)((ga >> 32) & 0x1FFFFFFull) | (2u << 30); // addr msbs + type=2
    i32x8 g1;
    g1[0] = (int)(2u << 16);   // data_size = 4B
    g1[1] = (int)(64u << 16);  // tensor_dim0 = 64
    g1[2] = (int)(64u << 16);  // tensor_dim1 = 64
    g1[3] = (int)(64u << 16);  // tile_dim0 = 64
    g1[4] = (int)(4u);         // tile_dim1 = 4
    g1[5] = 64;                // tensor_dim0_stride = 64
    g1[6] = 0;
    g1[7] = 0;
    i32x4 zz4 = { 0, 0, 0, 0 };
    i32x8 zz8 = { 0, 0, 0, 0, 0, 0, 0, 0 };
    if (threadIdx.x == 0) {
        __builtin_amdgcn_tensor_load_to_lds(g0, g1, zz4, zz4, zz8, 0);
    }
    __builtin_amdgcn_s_wait_tensorcnt(0);
    __builtin_amdgcn_s_cluster_barrier();   // NOP when not dispatched as a cluster
    __syncthreads();
    dst[threadIdx.x] = buf[threadIdx.x];
    dst[256 + threadIdx.x] = buf[256 + threadIdx.x];
}

// =====================================================================
// Host-side orchestration
// =====================================================================
extern "C" void kernel_launch(void* const* d_in, const int* in_sizes, int n_in,
                              void* d_out, int out_size, void* d_ws, size_t ws_size,
                              hipStream_t stream)
{
    (void)in_sizes; (void)n_in; (void)out_size; (void)ws_size;
    // ---- parameter leaves: JAX dict pytree flattens with sorted keys ----
    const float* input  = (const float*)d_in[0];
    const float* Wd     = (const float*)d_in[1];
    const float* Wf     = (const float*)d_in[2];
    const float* decW   = (const float*)d_in[3];
    const float* decB   = (const float*)d_in[4];
    const float* decG   = (const float*)d_in[5];
    const float* hl1W   = (const float*)d_in[6];
    const float* hl1B   = (const float*)d_in[7];
    const float* hl2W   = (const float*)d_in[8];
    const float* hl2B   = (const float*)d_in[9];
    const float* hlnB   = (const float*)d_in[10];
    const float* hlnG   = (const float*)d_in[11];
    const float* memW   = (const float*)d_in[12];
    const float* posW   = (const float*)d_in[13];
    const float* posB   = (const float*)d_in[14];
    const float* ppJoint= (const float*)d_in[15];
    const float* ppKvW  = (const float*)d_in[16];
    const float* ppL1W  = (const float*)d_in[17];
    const float* ppL1B  = (const float*)d_in[18];
    const float* ppL2W  = (const float*)d_in[19];
    const float* ppL2B  = (const float*)d_in[20];
    const float* ppL3W  = (const float*)d_in[21];
    const float* ppL3B  = (const float*)d_in[22];
    const float* ppL4W  = (const float*)d_in[23];
    const float* ppL4B  = (const float*)d_in[24];
    const float* ppLn1B = (const float*)d_in[25];
    const float* ppLn1G = (const float*)d_in[26];
    const float* ppLn2B = (const float*)d_in[27];
    const float* ppLn2G = (const float*)d_in[28];
    const float* ppOutW = (const float*)d_in[29];
    const float* ppOutB = (const float*)d_in[30];
    const float *trFF1W[5], *trFF1B[5], *trFF2W[5], *trFF2B[5], *trLn1B[5], *trLn1G[5],
                *trLn2B[5], *trLn2G[5], *trProjW[5], *trProjB[5], *trQkvW[5];
    for (int i = 0; i < 5; ++i) {
        int base = 31 + 11 * i;
        trFF1W[i]  = (const float*)d_in[base + 0];
        trFF1B[i]  = (const float*)d_in[base + 1];
        trFF2W[i]  = (const float*)d_in[base + 2];
        trFF2B[i]  = (const float*)d_in[base + 3];
        trLn1B[i]  = (const float*)d_in[base + 4];
        trLn1G[i]  = (const float*)d_in[base + 5];
        trLn2B[i]  = (const float*)d_in[base + 6];
        trLn2G[i]  = (const float*)d_in[base + 7];
        trProjW[i] = (const float*)d_in[base + 8];
        trProjB[i] = (const float*)d_in[base + 9];
        trQkvW[i]  = (const float*)d_in[base + 10];
    }

    // ---- workspace bump allocator ----
    size_t off = 0;
    auto alloc = [&](size_t bytes) -> void* {
        void* p = (char*)d_ws + off;
        off = (off + bytes + 255) & ~(size_t)255;
        return p;
    };
    auto fba = [&](size_t e) { return (float*)alloc(e * sizeof(float)); };
    auto hba = [&](size_t e) { return (__bf16*)alloc(e * sizeof(__bf16)); };
    auto iba = [&](size_t e) { return (int*)alloc(e * sizeof(int)); };

    // bf16 weights
    __bf16* w_joint = hba(P_ * DIM_);
    __bf16* w_kv    = hba(DIM_ * 256);
    __bf16* w_l1    = hba(6 * DIM_);
    __bf16* w_l2    = hba(DIM_ * DIM_);
    __bf16* w_l3    = hba(DIM_ * DIM_);
    __bf16* w_l4    = hba(DIM_ * 6);
    __bf16* w_out   = hba(DIM_ * DIM_);
    __bf16* w_pos   = hba(4 * DIM_);
    __bf16* w_mem   = hba(DIM_ * MEMN_);
    __bf16* w_dec   = hba(131 * DIM_);
    __bf16* w_hl1   = hba(DIM_ * 64);
    __bf16* w_hl2   = hba(64 * J_);
    __bf16 *w_qkv[5], *w_proj[5], *w_ff1[5], *w_ff2[5];
    for (int i = 0; i < 5; ++i) {
        w_qkv[i]  = hba(DIM_ * 384);
        w_proj[i] = hba(DIM_ * DIM_);
        w_ff1[i]  = hba(DIM_ * MLP_);
        w_ff2[i]  = hba(MLP_ * DIM_);
    }

    const int RP = FRAMES_ * N_;          // 32768 proposer tokens
    __bf16* flat_bf = hba((size_t)RP * 6);
    float*  h1      = fba((size_t)RP * DIM_);
    __bf16* h1bf    = hba((size_t)RP * DIM_);
    __bf16* h2bf    = hba((size_t)RP * DIM_);
    __bf16* kvbf    = hba((size_t)RP * 256);
    // shared attention scratch (proposer scores / transformer scores / mem scores)
    const size_t SCORE_E = (size_t)FRAMES_ * HP_ * P_ * N_;   // 8.4M f32 (largest)
    float*  scores  = fba(SCORE_E);
    __bf16* probs   = hba(SCORE_E);
    __bf16* opropbf = hba((size_t)FRAMES_ * P_ * DIM_);
    float*  o2      = fba((size_t)FRAMES_ * P_ * DIM_);
    __bf16* o2bf    = hba((size_t)FRAMES_ * P_ * DIM_);
    __bf16* o3bf    = hba((size_t)FRAMES_ * P_ * DIM_);
    float*  prop    = fba((size_t)FRAMES_ * P_ * 6);
    float*  full    = fba((size_t)FRAMES_ * NT_ * 6);
    int*    fpsidx  = iba((size_t)FRAMES_ * M_);
    float*  anchor  = fba((size_t)FRAMES_ * M_ * 3);
    float*  pooled  = fba((size_t)FRAMES_ * M_ * DIM_);
    float*  xyzts   = fba((size_t)RT_ * 4);
    __bf16* xyztsbf = hba((size_t)RT_ * 4);
    float*  xres    = fba((size_t)RT_ * DIM_);
    __bf16* lnbf    = hba((size_t)RT_ * DIM_);
    __bf16* qkvbf   = hba((size_t)RT_ * 384);
    __bf16* obf     = hba((size_t)RT_ * DIM_);
    float*  tmp     = fba((size_t)RT_ * DIM_);
    __bf16* ff1bf   = hba((size_t)RT_ * MLP_);
    __bf16* xbf     = hba((size_t)RT_ * DIM_);
    float*  memout  = fba((size_t)RT_ * DIM_);
    __bf16* deccat  = hba((size_t)B_ * NT_ * 131);
    float*  dech    = fba((size_t)B_ * NT_ * DIM_);
    float*  decout  = fba((size_t)B_ * NT_ * DIM_);
    __bf16* hhbf    = hba((size_t)B_ * NT_ * 64);
    float*  hlog    = fba((size_t)B_ * NT_ * J_);
    float*  tdmscr  = fba(1024);

    auto cvt = [&](const float* s, __bf16* dptr, size_t n) {
        convert_kernel<<<dim3((unsigned)((n + 255) / 256)), dim3(256), 0, stream>>>(s, dptr, (int)n);
    };
    // dispatch: fast kernel when every dimension is tile-exact and rows are 16B-aligned
    auto gemm = [&](const __bf16* A, int lda, long sA,
                    const __bf16* Bm, int ldb, long sB, int transB,
                    void* C, int ldc, long sC, int outBf,
                    int Mr, int Nc, int Kd, const float* bias, float alpha, int act, int batch) {
        dim3 grid((Nc + GTILE - 1) / GTILE, (Mr + GTILE - 1) / GTILE, batch);
        const bool fast = (Mr % GTILE == 0) && (Nc % GTILE == 0) && (Kd % GTILE == 0) &&
                          ((lda & 7) == 0) && ((ldb & 7) == 0);
        if (fast) {
            if (transB)
                gemm_wmma_fast<1><<<grid, dim3(128), 0, stream>>>(A, lda, sA, Bm, ldb, sB,
                    C, ldc, sC, Kd, bias, alpha, act, outBf);
            else
                gemm_wmma_fast<0><<<grid, dim3(128), 0, stream>>>(A, lda, sA, Bm, ldb, sB,
                    C, ldc, sC, Kd, bias, alpha, act, outBf);
        } else {
            gemm_wmma_gen<<<grid, dim3(128), 0, stream>>>(A, lda, sA, Bm, ldb, sB, C, ldc, sC,
                                                          Mr, Nc, Kd, bias, alpha, act, transB, outBf);
        }
    };

    // ---- weight conversions ----
    cvt(ppJoint, w_joint, P_ * DIM_);
    cvt(ppKvW,   w_kv,   DIM_ * 256);
    cvt(ppL1W,   w_l1,   6 * DIM_);
    cvt(ppL2W,   w_l2,   DIM_ * DIM_);
    cvt(ppL3W,   w_l3,   DIM_ * DIM_);
    cvt(ppL4W,   w_l4,   DIM_ * 6);
    cvt(ppOutW,  w_out,  DIM_ * DIM_);
    cvt(posW,    w_pos,  4 * DIM_);
    cvt(memW,    w_mem,  DIM_ * MEMN_);
    cvt(decW,    w_dec,  131 * DIM_);
    cvt(hl1W,    w_hl1,  DIM_ * 64);
    cvt(hl2W,    w_hl2,  64 * J_);
    for (int i = 0; i < 5; ++i) {
        cvt(trQkvW[i],  w_qkv[i],  DIM_ * 384);
        cvt(trProjW[i], w_proj[i], DIM_ * DIM_);
        cvt(trFF1W[i],  w_ff1[i],  DIM_ * MLP_);
        cvt(trFF2W[i],  w_ff2[i],  MLP_ * DIM_);
    }
    cvt(input, flat_bf, (size_t)RP * 6);

    // TDM warm-up of memory codebook (async-tensor path)
    tdm_warm_kernel<<<dim3(1), dim3(256), 0, stream>>>(memW, tdmscr);

    // ================= point proposer =================
    gemm(flat_bf, 6, 0, w_l1, DIM_, 0, 0, h1, DIM_, 0, 0, RP, DIM_, 6, ppL1B, 1.0f, 0, 1);
    ln_kernel<<<dim3(RP), dim3(128), 0, stream>>>(h1, ppLn1G, ppLn1B, h1bf, 1);
    gemm(h1bf, DIM_, 0, w_l2, DIM_, 0, 0, h2bf, DIM_, 0, 1, RP, DIM_, DIM_, ppL2B, 1.0f, 0, 1);
    gemm(h2bf, DIM_, 0, w_kv, 256, 0, 0, kvbf, 256, 0, 1, RP, 256, DIM_, nullptr, 1.0f, 0, 1);
    // cross-attention: batched over 32 frames, loop over 4 heads
    for (int hp = 0; hp < HP_; ++hp) {
        gemm(w_joint + hp * DHP_, DIM_, 0,
             kvbf + hp * DHP_, 256, (long)N_ * 256, 1,
             scores + (size_t)hp * P_ * N_, N_, (long)HP_ * P_ * N_, 0,
             P_, N_, DHP_, nullptr, 0.17677669529663687f, 0, FRAMES_);
    }
    softmax_kernel<<<dim3(FRAMES_ * HP_ * P_), dim3(256), 0, stream>>>(scores, probs, N_, N_);
    for (int hp = 0; hp < HP_; ++hp) {
        gemm(probs + (size_t)hp * P_ * N_, N_, (long)HP_ * P_ * N_,
             kvbf + 128 + hp * DHP_, 256, (long)N_ * 256, 0,
             opropbf + hp * DHP_, DIM_, (long)P_ * DIM_, 1,
             P_, DHP_, N_, nullptr, 1.0f, 0, FRAMES_);
    }
    gemm(opropbf, DIM_, 0, w_out, DIM_, 0, 0, o2, DIM_, 0, 0,
         FRAMES_ * P_, DIM_, DIM_, ppOutB, 1.0f, 0, 1);
    ln_kernel<<<dim3(FRAMES_ * P_), dim3(128), 0, stream>>>(o2, ppLn2G, ppLn2B, o2bf, 0);
    gemm(o2bf, DIM_, 0, w_l3, DIM_, 0, 0, o3bf, DIM_, 0, 1,
         FRAMES_ * P_, DIM_, DIM_, ppL3B, 1.0f, 1 /*gelu*/, 1);
    gemm(o3bf, DIM_, 0, w_l4, 6, 0, 0, prop, 6, 0, 0,
         FRAMES_ * P_, 6, DIM_, ppL4B, 1.0f, 0, 1);

    // ================= assemble, FPS, tube embedding =================
    {
        int total = FRAMES_ * NT_ * 6;
        build_full_kernel<<<dim3((total + 255) / 256), dim3(256), 0, stream>>>(input, prop, full);
    }
    fps_kernel<<<dim3(FRAMES_), dim3(256), 0, stream>>>(full, fpsidx);
    gather_anchor_kernel<<<dim3((FRAMES_ * M_ + 255) / 256), dim3(256), 0, stream>>>(full, fpsidx, anchor);
    for (int o = -1; o <= 1; ++o) {
        tube_kernel<<<dim3(M_, FRAMES_), dim3(128), 0, stream>>>(full, anchor, Wd, Wf, pooled, o, o == -1);
    }
    xyzts_kernel<<<dim3((FRAMES_ * M_ + 255) / 256), dim3(256), 0, stream>>>(anchor, xyzts);
    cvt(xyzts, xyztsbf, (size_t)RT_ * 4);
    gemm(xyztsbf, 4, 0, w_pos, DIM_, 0, 0, xres, DIM_, 0, 0, RT_, DIM_, 4, posB, 1.0f, 0, 1);
    add_relu_kernel<<<dim3((RT_ * DIM_ + 255) / 256), dim3(256), 0, stream>>>(xres, pooled, RT_ * DIM_);

    // ================= transformer (5 layers) =================
    const int TOK = L_ * M_;   // 1088 tokens per batch
    for (int ly = 0; ly < 5; ++ly) {
        ln_kernel<<<dim3(RT_), dim3(128), 0, stream>>>(xres, trLn1G[ly], trLn1B[ly], lnbf, 0);
        gemm(lnbf, DIM_, 0, w_qkv[ly], 384, 0, 0, qkvbf, 384, 0, 1, RT_, 384, DIM_, nullptr, 1.0f, 0, 1);
        for (int h = 0; h < HEADS_; ++h) {
            gemm(qkvbf + h * DH_, 384, (long)TOK * 384,
                 qkvbf + 128 + h * DH_, 384, (long)TOK * 384, 1,
                 scores, TOK, (long)TOK * TOK, 0,
                 TOK, TOK, DH_, nullptr, 0.25f, 0, B_);
            softmax_kernel<<<dim3(B_ * TOK), dim3(256), 0, stream>>>(scores, probs, TOK, TOK);
            gemm(probs, TOK, (long)TOK * TOK,
                 qkvbf + 256 + h * DH_, 384, (long)TOK * 384, 0,
                 obf + h * DH_, DIM_, (long)TOK * DIM_, 1,
                 TOK, DH_, TOK, nullptr, 1.0f, 0, B_);
        }
        gemm(obf, DIM_, 0, w_proj[ly], DIM_, 0, 0, tmp, DIM_, 0, 0,
             RT_, DIM_, DIM_, trProjB[ly], 1.0f, 0, 1);
        add_inplace_kernel<<<dim3((RT_ * DIM_ + 255) / 256), dim3(256), 0, stream>>>(xres, tmp, RT_ * DIM_);
        ln_kernel<<<dim3(RT_), dim3(128), 0, stream>>>(xres, trLn2G[ly], trLn2B[ly], lnbf, 0);
        gemm(lnbf, DIM_, 0, w_ff1[ly], MLP_, 0, 0, ff1bf, MLP_, 0, 1,
             RT_, MLP_, DIM_, trFF1B[ly], 1.0f, 1 /*gelu*/, 1);
        gemm(ff1bf, MLP_, 0, w_ff2[ly], DIM_, 0, 0, tmp, DIM_, 0, 0,
             RT_, DIM_, MLP_, trFF2B[ly], 1.0f, 0, 1);
        add_inplace_kernel<<<dim3((RT_ * DIM_ + 255) / 256), dim3(256), 0, stream>>>(xres, tmp, RT_ * DIM_);
    }

    // ================= memory attention =================
    cvt(xres, xbf, (size_t)RT_ * DIM_);
    gemm(xbf, DIM_, 0, w_mem, MEMN_, 0, 0, scores, MEMN_, 0, 0,
         RT_, MEMN_, DIM_, nullptr, 0.08838834764831845f, 0, 1);
    softmax_kernel<<<dim3(RT_), dim3(256), 0, stream>>>(scores, probs, MEMN_, MEMN_);
    gemm(probs, MEMN_, 0, w_mem, MEMN_, 0, 1 /*transB: mem^T*/, memout, DIM_, 0, 0,
         RT_, DIM_, MEMN_, nullptr, 1.0f, 0, 1);

    // ================= deconv (mid frame only) + head =================
    three_interp_kernel<<<dim3((B_ * NT_ + 255) / 256), dim3(256), 0, stream>>>(full, anchor, memout, deccat);
    gemm(deccat, 131, 0, w_dec, DIM_, 0, 0, dech, DIM_, 0, 0,
         B_ * NT_, DIM_, 131, nullptr, 1.0f, 0, 1);
    scalebias_relu_kernel<<<dim3((B_ * NT_ * DIM_ + 255) / 256), dim3(256), 0, stream>>>(
        dech, decG, decB, decout, B_ * NT_ * DIM_);
    ln_kernel<<<dim3(B_ * NT_), dim3(128), 0, stream>>>(decout, hlnG, hlnB, lnbf, 0);
    gemm(lnbf, DIM_, 0, w_hl1, 64, 0, 0, hhbf, 64, 0, 1,
         B_ * NT_, 64, DIM_, hl1B, 1.0f, 1 /*gelu*/, 1);
    gemm(hhbf, 64, 0, w_hl2, J_, 0, 0, hlog, J_, 0, 0,
         B_ * NT_, J_, 64, hl2B, 1.0f, 0, 1);
    head_final_kernel<<<dim3(B_ * J_), dim3(256), 0, stream>>>(hlog, full, (float*)d_out);
}
